// xLSTMModel_20349555048848
// MI455X (gfx1250) — matrix operand order repaired and checked by reading, not compiled, not run on hardware
//
#include <hip/hip_runtime.h>
#include <hip/hip_bf16.h>
#include <math.h>

#define DEV __device__ __forceinline__

typedef __bf16 bf16;
typedef __attribute__((ext_vector_type(16))) __bf16 v16bf;
typedef __attribute__((ext_vector_type(8)))  __bf16 bf16x8;
typedef __attribute__((ext_vector_type(8)))  float   v8f;
typedef int vint4 __attribute__((vector_size(4 * sizeof(int))));  // GCC-style, matches builtin

constexpr int cB   = 16;
constexpr int cS   = 512;
constexpr int cE   = 512;
constexpr int cD   = 1024;   // 2*E
constexpr int cNH  = 4;
constexpr int cTOK = cB * cS; // 8192 tokens

// ---------- async global->LDS (CDNA5 ASYNCcnt path), guarded ----------
#if defined(__has_builtin)
#if __has_builtin(__builtin_amdgcn_global_load_async_to_lds_b128)
#define HAS_ASYNC_LDS 1
#endif
#endif

#ifdef HAS_ASYNC_LDS
#if __has_builtin(__builtin_amdgcn_s_wait_asynccnt)
#define WAIT_ASYNC() __builtin_amdgcn_s_wait_asynccnt(0)
#else
#define WAIT_ASYNC() asm volatile("s_wait_asynccnt 0x0" ::: "memory")
#endif
#else
#define WAIT_ASYNC()
#endif

// copy 16 bytes global -> LDS (async DMA when available)
DEV void cp16(const bf16* __restrict__ g, bf16* l) {
#ifdef HAS_ASYNC_LDS
  __builtin_amdgcn_global_load_async_to_lds_b128(
      (__attribute__((address_space(1))) vint4*)(uintptr_t)g,
      (__attribute__((address_space(3))) vint4*)(uintptr_t)l, 0, 0);
#else
  *(bf16x8*)l = *(const bf16x8*)g;
#endif
}

// ---------- helpers ----------
DEV bf16 f2bf(float x) { return (bf16)x; }   // hardware RNE cvt
DEV v16bf mk16(bf16x8 lo, bf16x8 hi) {
  v16bf r;
#pragma unroll
  for (int i = 0; i < 8; i++) { r[i] = lo[i]; r[8 + i] = hi[i]; }
  return r;
}
DEV v8f wmma_bf(v16bf a, v16bf b, v8f c) {
  return __builtin_amdgcn_wmma_f32_16x16x32_bf16(false, a, false, b, (short)0, c, false, false);
}
DEV float sigm(float x)   { return 1.f / (1.f + __expf(-x)); }
DEV float logsig(float x) { return (x >= 0.f) ? -log1pf(__expf(-x)) : (x - log1pf(__expf(x))); }

// fp32 -> bf16 bulk convert
__global__ void k_cvt(const float* __restrict__ X, bf16* __restrict__ Y, int n) {
  int i = blockIdx.x * blockDim.x + threadIdx.x;
  if (i < n) Y[i] = f2bf(X[i]);
}

// =======================================================================
// WMMA GEMM, bf16 operands: C[m][n] = A[m][k] * W[n][k] (+bias, opt relu)
// M%64==0 (grid.x), N%64==0 (grid.y), K%32==0. 128 thr = 4 waves, 64x64.
// Tiles staged global->LDS via async DMA (no conversion in the hot loop).
// =======================================================================
__global__ __launch_bounds__(128)
void k_gemm(const bf16* __restrict__ A, int lda,
            const bf16* __restrict__ W, int ldw,
            float* __restrict__ C, int ldc,
            const float* __restrict__ bias, int Kdim, int act)
{
  __shared__ __align__(16) bf16 At[64][32];
  __shared__ __align__(16) bf16 Bt[64][32];
  const int row0 = blockIdx.x * 64, col0 = blockIdx.y * 64;
  const int t = threadIdx.x, wave = t >> 5, lane = t & 31;
  const int lrow = lane & 15, lhi = lane >> 4;

  v8f acc[4];
#pragma unroll
  for (int j = 0; j < 4; j++)
#pragma unroll
    for (int e = 0; e < 8; e++) acc[j][e] = 0.f;

  const int ch0 = t * 2;     // 2 16B-chunks per thread per matrix
  for (int k0 = 0; k0 < Kdim; k0 += 32) {
    __syncthreads();
#pragma unroll
    for (int i = 0; i < 2; i++) {
      int ch = ch0 + i, r = ch >> 2, off = (ch & 3) * 8;
      cp16(A + (size_t)(row0 + r) * lda + k0 + off, &At[r][off]);
      cp16(W + (size_t)(col0 + r) * ldw + k0 + off, &Bt[r][off]);
    }
    if (k0 + 32 < Kdim)
      __builtin_prefetch(A + (size_t)(row0 + (ch0 >> 2)) * lda + k0 + 32, 0, 0);
    WAIT_ASYNC();
    __syncthreads();

    const int ko = lhi * 8;
    bf16x8 a0 = *(const bf16x8*)&At[wave * 16 + lrow][ko];
    bf16x8 a1 = *(const bf16x8*)&At[wave * 16 + lrow][16 + ko];
    v16bf av = mk16(a0, a1);
#pragma unroll
    for (int j = 0; j < 4; j++) {
      const int n = j * 16 + lrow, kb = lhi * 8;
      bf16x8 b0 = *(const bf16x8*)&Bt[n][kb];
      bf16x8 b1 = *(const bf16x8*)&Bt[n][kb + 16];
      acc[j] = wmma_bf(av, mk16(b0, b1), acc[j]);
    }
  }

#pragma unroll
  for (int j = 0; j < 4; j++)
#pragma unroll
    for (int r = 0; r < 8; r++) {
      int m = row0 + wave * 16 + r + 8 * lhi;
      int n = col0 + j * 16 + lrow;
      float v = acc[j][r];
      if (bias) v += bias[n];
      if (act == 1) v = fmaxf(v, 0.f);
      C[(size_t)m * ldc + n] = v;
    }
}

// =======================================================================
// LayerNorm (dual fp32 + optional bf16 output). weight idx (row%wmod)*cols+c
// =======================================================================
__global__ __launch_bounds__(256)
void k_layernorm(const float* __restrict__ X, float* __restrict__ Y,
                 bf16* __restrict__ Ybf,
                 const float* __restrict__ w, int cols, int wmod)
{
  const int row = blockIdx.x, tt = threadIdx.x;
  const float* x = X + (size_t)row * cols;
  float* y = Y + (size_t)row * cols;
  const float* wr = w + (size_t)(row % wmod) * cols;
  __shared__ float rs[256], rs2[256];
  float s = 0.f, s2 = 0.f;
  for (int c = tt; c < cols; c += 256) { float v = x[c]; s += v; s2 += v * v; }
  rs[tt] = s; rs2[tt] = s2; __syncthreads();
  for (int o = 128; o > 0; o >>= 1) {
    if (tt < o) { rs[tt] += rs[tt + o]; rs2[tt] += rs2[tt + o]; }
    __syncthreads();
  }
  float mu = rs[0] / cols;
  float var = rs2[0] / cols - mu * mu;
  float inv = rsqrtf(var + 1e-5f);
  for (int c = tt; c < cols; c += 256) {
    float v = (x[c] - mu) * inv * wr[c];
    y[c] = v;
    if (Ybf) Ybf[(size_t)row * cols + c] = f2bf(v);
  }
}

// causal conv (K=4) + SiLU, dual fp32 + bf16 outputs
__global__ void k_conv_silu(const float* __restrict__ X, int ldx, int ch,
                            const float* __restrict__ w, const float* __restrict__ b,
                            float* __restrict__ Y, bf16* __restrict__ Ybf,
                            int ldy, int total)
{
  int i = blockIdx.x * blockDim.x + threadIdx.x;
  if (i >= total) return;
  int c = i % ch, tok = i / ch, s = tok % cS;
  float acc = b[c];
#pragma unroll
  for (int j = 0; j < 4; j++) {
    int sp = s - 3 + j;
    if (sp >= 0) acc += X[(size_t)(tok - 3 + j) * ldx + c] * w[c * 4 + j];
  }
  float v = acc * sigm(acc);
  Y[(size_t)tok * ldy + c] = v;
  Ybf[(size_t)tok * ldy + c] = f2bf(v);
}

// mLSTM headwise 4x4 block-diag; bf16 output (feeds WMMA attention)
__global__ void k_headwise4(const float* __restrict__ X, int ldx,
                            const float* __restrict__ W,
                            bf16* __restrict__ Y, int total)
{
  int i = blockIdx.x * blockDim.x + threadIdx.x;
  if (i >= total) return;
  int c = i % cD, tok = i / cD, nb = c >> 2, o = c & 3;
  const float* xb = X + (size_t)tok * ldx + (nb << 2);
  const float* wb = W + nb * 16 + o * 4;
  Y[(size_t)tok * cD + c] = f2bf(xb[0]*wb[0] + xb[1]*wb[1] + xb[2]*wb[2] + xb[3]*wb[3]);
}

// mLSTM input/forget gates: 1 block per token, wave = head
__global__ __launch_bounds__(128)
void k_mgates(const bf16* __restrict__ Q, const bf16* __restrict__ K,
              const bf16* __restrict__ V,
              const float* __restrict__ igw, const float* __restrict__ igb,
              const float* __restrict__ fgw, const float* __restrict__ fgb,
              float* __restrict__ IG, float* __restrict__ FG)
{
  const int tok = blockIdx.x, h = threadIdx.x >> 5, lane = threadIdx.x & 31;
  const float* iw = igw + h * 3072;
  const float* fw = fgw + h * 3072;
  float si = 0.f, sf = 0.f;
  for (int j = lane; j < cD; j += 32) {
    float q = (float)Q[(size_t)tok * cD + j];
    float k = (float)K[(size_t)tok * cD + j];
    float v = (float)V[(size_t)tok * cD + j];
    si += q * iw[j] + k * iw[cD + j] + v * iw[2 * cD + j];
    sf += q * fw[j] + k * fw[cD + j] + v * fw[2 * cD + j];
  }
#pragma unroll
  for (int o = 16; o > 0; o >>= 1) { si += __shfl_xor(si, o); sf += __shfl_xor(sf, o); }
  if (lane == 0) {
    int b = tok / cS, s = tok % cS;
    IG[(size_t)(b * cNH + h) * cS + s] = si + igb[h];
    FG[(size_t)(b * cNH + h) * cS + s] = sf + fgb[h];
  }
}

// per-(b,h) serial prefix: a[t]=ig[t]-Lcum[t], cm[s]=prefixmax(a), m[s]=Lcum[s]+cm[s]
__global__ void k_prescan(const float* __restrict__ IG, const float* __restrict__ FG,
                          float* __restrict__ Aa, float* __restrict__ Cm, float* __restrict__ Mm)
{
  int bh = threadIdx.x;
  if (bh >= cB * cNH) return;
  const float* ig = IG + (size_t)bh * cS;
  const float* fg = FG + (size_t)bh * cS;
  float L = 0.f, cm = -1e30f;
  for (int s = 0; s < cS; s++) {
    L += logsig(fg[s]);
    float a = ig[s] - L;
    cm = fmaxf(cm, a);
    Aa[(size_t)bh * cS + s] = a;
    Cm[(size_t)bh * cS + s] = cm;
    Mm[(size_t)bh * cS + s] = L + cm;
  }
}

// =======================================================================
// mLSTM parallel attention, flash-style, bf16 operands.
// grid (b*nh=64, 8); 128 thr = 4 waves; wave = one 16-row query tile.
// K tiles staged to LDS via async DMA; V staged transposed manually.
// =======================================================================
__global__ __launch_bounds__(128)
void k_attn(const bf16* __restrict__ Q, const bf16* __restrict__ K,
            const bf16* __restrict__ V,
            const float* __restrict__ Aa, const float* __restrict__ Cm,
            const float* __restrict__ Mm, float* __restrict__ O)
{
  __shared__ __align__(16) bf16 Kt[32][256];
  __shared__ __align__(16) bf16 VtT[256][32];
  __shared__ __align__(16) bf16 Cpack[4][16][32];

  const int bh = blockIdx.x, g = blockIdx.y;
  const int b = bh >> 2, h = bh & 3;
  const int t = threadIdx.x, wave = t >> 5, lane = t & 31;
  const int lrow = lane & 15, lhi = lane >> 4;
  const int qb = g * 64 + wave * 16;
  const size_t base = ((size_t)b * cS) * cD + (size_t)h * 256;

  // preload q fragments (16 rows x 256) straight from global bf16
  v16bf qf[8];
  {
    const bf16* qr = Q + base + (size_t)(qb + lrow) * cD;
#pragma unroll
    for (int kc = 0; kc < 8; kc++) {
      int o0 = kc * 32 + lhi * 8;
      bf16x8 a0 = *(const bf16x8*)(qr + o0);
      bf16x8 a1 = *(const bf16x8*)(qr + o0 + 16);
      qf[kc] = mk16(a0, a1);
    }
  }
  float cmr[8], mr[8], rsum[8];
  {
    const float* cmb = Cm + (size_t)bh * cS;
    const float* mb  = Mm + (size_t)bh * cS;
#pragma unroll
    for (int r = 0; r < 8; r++) {
      int s = qb + r + 8 * lhi;
      cmr[r] = cmb[s]; mr[r] = mb[s]; rsum[r] = 0.f;
    }
  }
  v8f acc[16];
#pragma unroll
  for (int j = 0; j < 16; j++)
#pragma unroll
    for (int e = 0; e < 8; e++) acc[j][e] = 0.f;

  const float* ab = Aa + (size_t)bh * cS;
  const int ktmax = 2 * g + 1;

  for (int kt = 0; kt <= ktmax; kt++) {
    __syncthreads();
    { // K tile (32x256): 1024 16B chunks, 8 per thread, async DMA to LDS
      const bf16* kb0 = K + base + (size_t)(kt * 32) * cD;
#pragma unroll
      for (int i = 0; i < 8; i++) {
        int ch = t * 8 + i, r = ch >> 5, off = (ch & 31) * 8;
        cp16(kb0 + (size_t)r * cD + off, &Kt[r][off]);
      }
      // V tile transposed: vector-load rows, scatter 2B stores
      int r = t >> 2, c0 = (t & 3) * 64;
      const bf16* vr = V + base + (size_t)(kt * 32 + r) * cD + c0;
#pragma unroll
      for (int j = 0; j < 8; j++) {
        bf16x8 vv = *(const bf16x8*)(vr + j * 8);
#pragma unroll
        for (int e = 0; e < 8; e++) VtT[c0 + j * 8 + e][r] = vv[e];
      }
    }
    WAIT_ASYNC();
    __syncthreads();

    const bool active = (kt * 32) <= (qb + 15);
    if (active) {
      // scores 16x32 = Q(16x256) * K(32x256)^T
      v8f sa[2];
#pragma unroll
      for (int j = 0; j < 2; j++) {
#pragma unroll
        for (int e = 0; e < 8; e++) sa[j][e] = 0.f;
        const int n = j * 16 + lrow, kb = lhi * 8;
#pragma unroll
        for (int kc = 0; kc < 8; kc++) {
          bf16x8 b0 = *(const bf16x8*)&Kt[n][kc * 32 + kb];
          bf16x8 b1 = *(const bf16x8*)&Kt[n][kc * 32 + kb + 16];
          sa[j] = wmma_bf(qf[kc], mk16(b0, b1), sa[j]);
        }
      }
      // decay, causal mask, rowsum; repack to bf16 A-layout via LDS
#pragma unroll
      for (int j = 0; j < 2; j++) {
        int tg = kt * 32 + j * 16 + lrow;
        float at = ab[tg];
#pragma unroll
        for (int r = 0; r < 8; r++) {
          int srow = qb + r + 8 * lhi;
          float v = 0.f;
          if (tg <= srow) v = sa[j][r] * 0.0625f * __expf(at - cmr[r]);
          rsum[r] += v;
          Cpack[wave][r + 8 * lhi][j * 16 + lrow] = f2bf(v);
        }
      }
    }
    // intra-wave LDS RAW on Cpack
    asm volatile("s_wait_dscnt 0x0" ::: "memory");
    if (active) {
      const int ko = lhi * 8;
      bf16x8 a0 = *(const bf16x8*)&Cpack[wave][lrow][ko];
      bf16x8 a1 = *(const bf16x8*)&Cpack[wave][lrow][16 + ko];
      v16bf av = mk16(a0, a1);
#pragma unroll
      for (int j2 = 0; j2 < 16; j2++) {
        const int d = j2 * 16 + lrow, kb = lhi * 8;
        bf16x8 b0 = *(const bf16x8*)&VtT[d][kb];
        bf16x8 b1 = *(const bf16x8*)&VtT[d][kb + 16];
        acc[j2] = wmma_bf(av, mk16(b0, b1), acc[j2]);
      }
    }
  }

  // reduce rowsum across the 16 lanes holding each row
#pragma unroll
  for (int r = 0; r < 8; r++) {
    float v = rsum[r];
    v += __shfl_xor(v, 1); v += __shfl_xor(v, 2); v += __shfl_xor(v, 4); v += __shfl_xor(v, 8);
    rsum[r] = v;
  }
#pragma unroll
  for (int r = 0; r < 8; r++) {
    int srow = qb + r + 8 * lhi;
    float norm = fmaxf(fabsf(rsum[r]), __expf(-mr[r])) + 1e-6f;
    float inv = 1.f / norm;
#pragma unroll
    for (int j2 = 0; j2 < 16; j2++)
      O[base + (size_t)srow * cD + j2 * 16 + lrow] = acc[j2][r] * inv;
  }
}

// t = (mhln + skip*xc) * silu(z)  -> bf16 (feeds proj_down GEMM)
__global__ void k_mcombine(const float* __restrict__ HA, const float* __restrict__ XC,
                           const float* __restrict__ XIN, const float* __restrict__ skip,
                           bf16* __restrict__ T, int total)
{
  int i = blockIdx.x * blockDim.x + threadIdx.x;
  if (i >= total) return;
  int c = i % cD, tok = i / cD;
  float z = XIN[(size_t)tok * 2048 + cD + c];
  T[i] = f2bf((HA[i] + skip[c] * XC[i]) * z * sigm(z));
}

__global__ void k_add(float* __restrict__ Y, const float* __restrict__ X, int n)
{
  int i = blockIdx.x * blockDim.x + threadIdx.x;
  if (i < n) Y[i] += X[i];
}

// t = gelu_exact(u[:,:704]) * u[:,704:] -> bf16 (feeds ff_down GEMM)
__global__ void k_geglu(const float* __restrict__ U, bf16* __restrict__ T, int total)
{
  int i = blockIdx.x * blockDim.x + threadIdx.x;
  if (i >= total) return;
  int j = i % 704, tok = i / 704;
  float u  = U[(size_t)tok * 1408 + j];
  float up = U[(size_t)tok * 1408 + 704 + j];
  T[i] = f2bf(0.5f * u * (1.f + erff(u * 0.70710678118f)) * up);
}

// sLSTM recurrent scan: one 128-thread block per (b,h); thread = channel o
__global__ __launch_bounds__(128)
void k_sscan(const float* __restrict__ G, const float* __restrict__ RW,
             const float* __restrict__ bias, float* __restrict__ Y)
{
  const int bh = blockIdx.x, b = bh >> 2, h = bh & 3, o = threadIdx.x;
  __shared__ float ylds[128];
  float c = 0.f, n = 0.f, mm = 0.f;
  ylds[o] = 0.f; __syncthreads();
  const float* rw0 = RW + ((size_t)((0 * 4 + h) * 128 + o)) * 128;
  const float* rw1 = RW + ((size_t)((1 * 4 + h) * 128 + o)) * 128;
  const float* rw2 = RW + ((size_t)((2 * 4 + h) * 128 + o)) * 128;
  const float* rw3 = RW + ((size_t)((3 * 4 + h) * 128 + o)) * 128;
  const float bb0 = bias[(0 * 4 + h) * 128 + o], bb1 = bias[(1 * 4 + h) * 128 + o];
  const float bb2 = bias[(2 * 4 + h) * 128 + o], bb3 = bias[(3 * 4 + h) * 128 + o];
  const size_t gst = (size_t)cTOK * cE;
  for (int s = 0; s < cS; s++) {
    float r0 = bb0, r1 = bb1, r2 = bb2, r3 = bb3;
#pragma unroll 4
    for (int d = 0; d < 128; d++) {
      float y = ylds[d];
      r0 += rw0[d] * y; r1 += rw1[d] * y; r2 += rw2[d] * y; r3 += rw3[d] * y;
    }
    size_t gi = (size_t)(b * cS + s) * cE + h * 128 + o;
    float iraw = G[gi] + r0, fraw = G[gst + gi] + r1;
    float zraw = G[2 * gst + gi] + r2, oraw = G[3 * gst + gi] + r3;
    float lfm  = mm + logsig(fraw);
    float mnew = fmaxf(iraw, lfm);
    float ig = __expf(iraw - mnew), fg = __expf(lfm - mnew);
    float cn = fg * c + ig * tanhf(zraw);
    float nn = fg * n + ig;
    float yn = sigm(oraw) * cn / nn;
    c = cn; n = nn; mm = mnew;
    Y[gi] = yn;
    __syncthreads();
    ylds[o] = yn;
    __syncthreads();
  }
}

// final: gather feats at x_len-1, post-LN, fc1+relu, fc2, sigmoid
__global__ __launch_bounds__(256)
void k_head(const float* __restrict__ H, const int* __restrict__ xlen,
            const float* __restrict__ plw,
            const float* __restrict__ w1, const float* __restrict__ b1,
            const float* __restrict__ w2, const float* __restrict__ b2,
            float* __restrict__ out)
{
  const int b = blockIdx.x, tt = threadIdx.x;
  __shared__ float feats[cE];
  __shared__ float rs[256], rs2[256];
  const int idx = xlen[b] - 1;
  const float* x = H + (size_t)(b * cS + idx) * cE;
  float s = 0.f, s2 = 0.f;
  for (int c = tt; c < cE; c += 256) { float v = x[c]; s += v; s2 += v * v; }
  rs[tt] = s; rs2[tt] = s2; __syncthreads();
  for (int o = 128; o > 0; o >>= 1) {
    if (tt < o) { rs[tt] += rs[tt + o]; rs2[tt] += rs2[tt + o]; }
    __syncthreads();
  }
  float mu = rs[0] / cE, var = rs2[0] / cE - mu * mu;
  float inv = rsqrtf(var + 1e-5f);
  for (int c = tt; c < cE; c += 256) {
    float v = (x[c] - mu) * inv * plw[c];
    feats[c] = v;
    out[16 + b * cE + c] = v;
  }
  __syncthreads();
  float a = b1[tt];
  const float* wr = w1 + (size_t)tt * cE;
  for (int c = 0; c < cE; c++) a += wr[c] * feats[c];
  a = fmaxf(a, 0.f);
  rs[tt] = a * w2[tt];
  __syncthreads();
  for (int o = 128; o > 0; o >>= 1) {
    if (tt < o) rs[tt] += rs[tt + o];
    __syncthreads();
  }
  if (tt == 0) out[b] = sigm(rs[0] + b2[0]);
}

// =======================================================================
extern "C" void kernel_launch(void* const* d_in, const int* in_sizes, int n_in,
                              void* d_out, int out_size, void* d_ws, size_t ws_size,
                              hipStream_t stream)
{
  const float* x          = (const float*)d_in[0];
  const int*   x_len      = (const int*)d_in[1];
  const float* w_inp      = (const float*)d_in[2];
  const float* m_ln_w     = (const float*)d_in[3];
  const float* m_proj_up  = (const float*)d_in[4];
  const float* m_conv_w   = (const float*)d_in[5];
  const float* m_conv_b   = (const float*)d_in[6];
  const float* m_q_w      = (const float*)d_in[7];
  const float* m_k_w      = (const float*)d_in[8];
  const float* m_v_w      = (const float*)d_in[9];
  const float* m_ig_w     = (const float*)d_in[10];
  const float* m_ig_b     = (const float*)d_in[11];
  const float* m_fg_w     = (const float*)d_in[12];
  const float* m_fg_b     = (const float*)d_in[13];
  const float* m_mhln_w   = (const float*)d_in[14];
  const float* m_skip     = (const float*)d_in[15];
  const float* m_proj_down= (const float*)d_in[16];
  const float* s_ln_w     = (const float*)d_in[17];
  const float* s_conv_w   = (const float*)d_in[18];
  const float* s_conv_b   = (const float*)d_in[19];
  const float* s_gate_w   = (const float*)d_in[20];
  const float* s_rec_w    = (const float*)d_in[21];
  const float* s_bias     = (const float*)d_in[22];
  const float* s_mhln_w   = (const float*)d_in[23];
  const float* s_ln2_w    = (const float*)d_in[24];
  const float* s_ff_up    = (const float*)d_in[25];
  const float* s_ff_down  = (const float*)d_in[26];
  const float* post_ln_w  = (const float*)d_in[27];
  const float* w_fc1      = (const float*)d_in[28];
  const float* b_fc1      = (const float*)d_in[29];
  const float* w_fc2      = (const float*)d_in[30];
  const float* b_fc2      = (const float*)d_in[31];

  const size_t TOK = cTOK;

  // ---- workspace arena: fp32 region then bf16 region (~285 MB peak) ----
  float* ws   = (float*)d_ws;
  float* b_h   = ws;                       // TOK*512
  float* b_xn  = b_h   + TOK * 512;        // TOK*512
  float* b_xin = b_xn  + TOK * 512;        // TOK*2048 (sLSTM gates / ff u)
  float* b_xc  = b_xin + TOK * 2048;       // TOK*1024
  float* b_ha  = b_xc  + TOK * 1024;       // TOK*1024
  float* b_ig  = b_ha  + TOK * 1024;       // 64*512
  float* b_fg  = b_ig  + 64 * 512;
  float* b_a   = b_fg  + 64 * 512;
  float* b_cm  = b_a   + 64 * 512;
  float* b_m   = b_cm  + 64 * 512;

  bf16* bfa    = (bf16*)(b_m + 64 * 512);
  bf16* xn_bf  = bfa;                      // TOK*512
  bf16* xc_bf  = xn_bf + TOK * 512;        // TOK*1024
  bf16* t_bf   = xc_bf + TOK * 1024;       // TOK*1024
  bf16* q_bf   = t_bf  + TOK * 1024;       // TOK*1024
  bf16* k_bf   = q_bf  + TOK * 1024;
  bf16* v_bf   = k_bf  + TOK * 1024;
  bf16* x_bf   = v_bf  + TOK * 1024;       // TOK*768
  bf16* wb_inp = x_bf  + TOK * 768;        // 512*768
  bf16* wb_pu  = wb_inp+ 512 * 768;        // 3*2048*512
  bf16* wb_pd  = wb_pu + (size_t)3 * 2048 * 512;  // 3*512*1024
  bf16* wb_gw  = wb_pd + (size_t)3 * 512 * 1024;  // 4*4*128*128
  bf16* wb_ffu = wb_gw + 4 * 4 * 128 * 128;       // 1408*512
  bf16* wb_ffd = wb_ffu+ 1408 * 512;              // 512*704

  const int EW = 256;
  auto cdiv = [](int a, int b) { return (a + b - 1) / b; };
  auto cvt = [&](const float* src, bf16* dst, int n) {
    k_cvt<<<cdiv(n, EW), EW, 0, stream>>>(src, dst, n);
  };

  // one-time (per launch) bf16 conversion of WMMA operands
  cvt(x,           x_bf,   (int)(TOK * 768));
  cvt(w_inp,       wb_inp, 512 * 768);
  cvt(m_proj_up,   wb_pu,  3 * 2048 * 512);
  cvt(m_proj_down, wb_pd,  3 * 512 * 1024);
  cvt(s_gate_w,    wb_gw,  4 * 4 * 128 * 128);
  cvt(s_ff_up,     wb_ffu, 1408 * 512);
  cvt(s_ff_down,   wb_ffd, 512 * 704);

  // input projection: h = x @ w_inp^T
  k_gemm<<<dim3(TOK / 64, 512 / 64), 128, 0, stream>>>(
      x_bf, 768, wb_inp, 768, b_h, 512, nullptr, 768, 0);

  int mi = 0;
  for (int blk = 0; blk < 4; blk++) {
    if (blk == 1) {
      // ---------------- sLSTM block ----------------
      k_layernorm<<<TOK, 256, 0, stream>>>(b_h, b_xn, xn_bf, s_ln_w, 512, 1);
      k_conv_silu<<<cdiv(TOK * 512, EW), EW, 0, stream>>>(
          b_xn, 512, 512, s_conv_w, s_conv_b, b_xc, xc_bf, 512, TOK * 512);
      for (int g = 0; g < 4; g++) {
        const bf16* src = (g < 2) ? xc_bf : xn_bf;
        for (int hh = 0; hh < 4; hh++) {
          k_gemm<<<dim3(TOK / 64, 2), 128, 0, stream>>>(
              src + hh * 128, 512,
              wb_gw + (size_t)(g * 4 + hh) * 128 * 128, 128,
              b_xin + (size_t)g * TOK * 512 + hh * 128, 512,
              nullptr, 128, 0);
        }
      }
      k_sscan<<<64, 128, 0, stream>>>(b_xin, s_rec_w, s_bias, b_ha);
      k_layernorm<<<TOK * 4, 256, 0, stream>>>(b_ha, b_ha, (bf16*)nullptr, s_mhln_w, 128, 4);
      k_add<<<cdiv(TOK * 512, EW), EW, 0, stream>>>(b_h, b_ha, TOK * 512);
      k_layernorm<<<TOK, 256, 0, stream>>>(b_h, b_xn, xn_bf, s_ln2_w, 512, 1);
      k_gemm<<<dim3(TOK / 64, 1408 / 64), 128, 0, stream>>>(
          xn_bf, 512, wb_ffu, 512, b_xin, 1408, nullptr, 512, 0);
      k_geglu<<<cdiv(TOK * 704, EW), EW, 0, stream>>>(b_xin, t_bf, TOK * 704);
      k_gemm<<<dim3(TOK / 64, 512 / 64), 128, 0, stream>>>(
          t_bf, 704, wb_ffd, 704, b_xc, 512, nullptr, 704, 0);
      k_add<<<cdiv(TOK * 512, EW), EW, 0, stream>>>(b_h, b_xc, TOK * 512);
    } else {
      // ---------------- mLSTM block ----------------
      k_layernorm<<<TOK, 256, 0, stream>>>(b_h, b_xn, xn_bf, m_ln_w + mi * 512, 512, 1);
      k_gemm<<<dim3(TOK / 64, 2048 / 64), 128, 0, stream>>>(
          xn_bf, 512, wb_pu + (size_t)mi * 2048 * 512, 512,
          b_xin, 2048, nullptr, 512, 0);
      k_conv_silu<<<cdiv(TOK * 1024, EW), EW, 0, stream>>>(
          b_xin, 2048, 1024, m_conv_w + mi * 1024 * 4, m_conv_b + mi * 1024,
          b_xc, xc_bf, 1024, TOK * 1024);
      k_headwise4<<<cdiv(TOK * 1024, EW), EW, 0, stream>>>(
          b_xc, 1024, m_q_w + mi * 256 * 16, q_bf, TOK * 1024);
      k_headwise4<<<cdiv(TOK * 1024, EW), EW, 0, stream>>>(
          b_xc, 1024, m_k_w + mi * 256 * 16, k_bf, TOK * 1024);
      k_headwise4<<<cdiv(TOK * 1024, EW), EW, 0, stream>>>(
          b_xin, 2048, m_v_w + mi * 256 * 16, v_bf, TOK * 1024);
      k_mgates<<<TOK, 128, 0, stream>>>(
          q_bf, k_bf, v_bf,
          m_ig_w + mi * 4 * 3072, m_ig_b + mi * 4,
          m_fg_w + mi * 4 * 3072, m_fg_b + mi * 4, b_ig, b_fg);
      k_prescan<<<1, 64, 0, stream>>>(b_ig, b_fg, b_a, b_cm, b_m);
      k_attn<<<dim3(64, 8), 128, 0, stream>>>(q_bf, k_bf, v_bf, b_a, b_cm, b_m, b_ha);
      k_layernorm<<<TOK * 4, 256, 0, stream>>>(b_ha, b_ha, (bf16*)nullptr,
                                               m_mhln_w + mi * 1024, 256, 4);
      k_mcombine<<<cdiv(TOK * 1024, EW), EW, 0, stream>>>(
          b_ha, b_xc, b_xin, m_skip + mi * 1024, t_bf, TOK * 1024);
      k_gemm<<<dim3(TOK / 64, 512 / 64), 128, 0, stream>>>(
          t_bf, 1024, wb_pd + (size_t)mi * 512 * 1024, 1024,
          b_xn, 512, nullptr, 1024, 0);
      k_add<<<cdiv(TOK * 512, EW), EW, 0, stream>>>(b_h, b_xn, TOK * 512);
      mi++;
    }
  }

  k_head<<<16, 256, 0, stream>>>(b_h, x_len, post_ln_w, w_fc1, b_fc1, w_fc2, b_fc2,
                                 (float*)d_out);
}
